// GCN_19000935318233
// MI455X (gfx1250) — compile-verified, matrix operand order checked by reading, not run on hardware
//
#include <hip/hip_runtime.h>
#include <hip/hip_bf16.h>

typedef __attribute__((ext_vector_type(16))) _Float16 v16h;
typedef __attribute__((ext_vector_type(8)))  _Float16 v8h;
typedef __attribute__((ext_vector_type(8)))  float    v8f;

#define F_IN   128
#define TYPES  64
#define ALPHA  0.85f
#define WT_STRIDE 136   // 128 + 8 pad halves -> 272B row, 16B aligned

// ---------------------------------------------------------------- utilities
__global__ void k_zero(float* __restrict__ p, long long n) {
  long long i = blockIdx.x * (long long)blockDim.x + threadIdx.x;
  if (i < n) p[i] = 0.0f;
}

__global__ void k_copy(const float* __restrict__ s, float* __restrict__ d, long long n) {
  long long i = blockIdx.x * (long long)blockDim.x + threadIdx.x;
  if (i < n) d[i] = s[i];
}

// ------------------------------------------------ phase 1: SAGE aggregation
__global__ void k_deg(const int* __restrict__ dst, float* __restrict__ deg, long long e) {
  long long i = blockIdx.x * (long long)blockDim.x + threadIdx.x;
  if (i < e) atomicAdd(&deg[dst[i]], 1.0f);
}

// one thread per (edge, feature): 128-thread groups touch one 512B row -> coalesced
__global__ void k_msg_scatter(const int* __restrict__ src, const int* __restrict__ dst,
                              const float* __restrict__ x, float* __restrict__ msg,
                              long long e) {
  long long gid = blockIdx.x * (long long)blockDim.x + threadIdx.x;
  if (gid >= e * (long long)F_IN) return;
  int       f = (int)(gid & (F_IN - 1));
  long long i = gid >> 7;
  long long s = src[i], d = dst[i];
  atomicAdd(&msg[d * F_IN + f], x[s * F_IN + f]);
}

// ------------------------------- phase 2: dual GEMM + normalize + sigmoid (WMMA)
// Weights staged once per block into LDS as f16, transposed to B-operand layout:
// WT[col][k] so each lane's 8-half K-run is one contiguous 16B ds_load_b128.
// One wave computes 16 rows x 64 cols; 4 K-steps x 4 col-tiles x 2 GEMMs = 32 WMMA.
__global__ __launch_bounds__(256) void k_sage_wmma(
    const float* __restrict__ msg, const float* __restrict__ deg,
    const float* __restrict__ x,   const float* __restrict__ Wl,
    const float* __restrict__ bl,  const float* __restrict__ Wr,
    float* __restrict__ h, int n)
{
  __shared__ _Float16 WlT[TYPES * WT_STRIDE];
  __shared__ _Float16 WrT[TYPES * WT_STRIDE];

  // cooperative stage: coalesced global reads of W rows, f32->f16, transpose
  for (int idx = threadIdx.x; idx < F_IN * TYPES; idx += 256) {
    const int k   = idx >> 6;        // 0..127
    const int col = idx & 63;        // consecutive threads -> consecutive cols
    WlT[col * WT_STRIDE + k] = (_Float16)Wl[idx];
    WrT[col * WT_STRIDE + k] = (_Float16)Wr[idx];
  }
  __syncthreads();                   // before any wave-uniform early return

  const int wid  = (int)((blockIdx.x * (long long)blockDim.x + threadIdx.x) >> 5);
  const int lane = threadIdx.x & 31;
  const int row0 = wid * 16;
  if (row0 >= n) return;             // wave-uniform: EXEC all-1s for WMMA
  const int  m     = lane & 15;      // A: M row / B,C: N col within tile
  const int  khalf = lane >> 4;      // K-half selector per ISA 16-bit layout
  const int  row   = row0 + m;
  const bool valid = row < n;        // N % 16 == 0 here, but keep guard
  float dinv = 0.0f;
  if (valid) dinv = 1.0f / fmaxf(deg[row], 1.0f);
  const float* mrow = msg + (long long)row * F_IN;
  const float* xrow = x   + (long long)row * F_IN;

  v8f acc[4] = {v8f{}, v8f{}, v8f{}, v8f{}};

  #pragma unroll
  for (int k0 = 0; k0 < F_IN; k0 += 32) {
    // A: lane<16 covers K={kb0..kb0+7, kb1..kb1+7}; both runs 32B-aligned
    const int kb0 = k0 + khalf * 8;
    const int kb1 = k0 + 16 + khalf * 8;
    float4 a0{}, a1{}, a2{}, a3{}, g0{}, g1{}, g2{}, g3{};
    if (valid) {
      a0 = *(const float4*)(mrow + kb0); a1 = *(const float4*)(mrow + kb0 + 4);
      a2 = *(const float4*)(mrow + kb1); a3 = *(const float4*)(mrow + kb1 + 4);
      g0 = *(const float4*)(xrow + kb0); g1 = *(const float4*)(xrow + kb0 + 4);
      g2 = *(const float4*)(xrow + kb1); g3 = *(const float4*)(xrow + kb1 + 4);
    }
    v16h am, ax;
    am[0]=(_Float16)(a0.x*dinv); am[1]=(_Float16)(a0.y*dinv);
    am[2]=(_Float16)(a0.z*dinv); am[3]=(_Float16)(a0.w*dinv);
    am[4]=(_Float16)(a1.x*dinv); am[5]=(_Float16)(a1.y*dinv);
    am[6]=(_Float16)(a1.z*dinv); am[7]=(_Float16)(a1.w*dinv);
    am[8]=(_Float16)(a2.x*dinv); am[9]=(_Float16)(a2.y*dinv);
    am[10]=(_Float16)(a2.z*dinv); am[11]=(_Float16)(a2.w*dinv);
    am[12]=(_Float16)(a3.x*dinv); am[13]=(_Float16)(a3.y*dinv);
    am[14]=(_Float16)(a3.z*dinv); am[15]=(_Float16)(a3.w*dinv);
    ax[0]=(_Float16)g0.x; ax[1]=(_Float16)g0.y; ax[2]=(_Float16)g0.z; ax[3]=(_Float16)g0.w;
    ax[4]=(_Float16)g1.x; ax[5]=(_Float16)g1.y; ax[6]=(_Float16)g1.z; ax[7]=(_Float16)g1.w;
    ax[8]=(_Float16)g2.x; ax[9]=(_Float16)g2.y; ax[10]=(_Float16)g2.z; ax[11]=(_Float16)g2.w;
    ax[12]=(_Float16)g3.x; ax[13]=(_Float16)g3.y; ax[14]=(_Float16)g3.z; ax[15]=(_Float16)g3.w;

    #pragma unroll
    for (int ct = 0; ct < 4; ++ct) {
      const int col = ct * 16 + m;   // B: lane holds column `col`
      // two contiguous 8-half runs per weight matrix -> ds_load_b128 each
      const v8h l0 = *(const v8h*)&WlT[col * WT_STRIDE + kb0];
      const v8h l1 = *(const v8h*)&WlT[col * WT_STRIDE + kb1];
      const v8h r0 = *(const v8h*)&WrT[col * WT_STRIDE + kb0];
      const v8h r1 = *(const v8h*)&WrT[col * WT_STRIDE + kb1];
      v16h bL, bR;
      #pragma unroll
      for (int j = 0; j < 8; ++j) {
        bL[j] = l0[j]; bL[8 + j] = l1[j];
        bR[j] = r0[j]; bR[8 + j] = r1[j];
      }
      acc[ct] = __builtin_amdgcn_wmma_f32_16x16x32_f16(
          false, am, false, bL, (short)0, acc[ct], false, false);
      acc[ct] = __builtin_amdgcn_wmma_f32_16x16x32_f16(
          false, ax, false, bR, (short)0, acc[ct], false, false);
    }
  }

  float bias[4];
  #pragma unroll
  for (int ct = 0; ct < 4; ++ct) bias[ct] = bl[ct * 16 + m];

  // C layout: acc[ct][v] is element (row0 + v + 8*khalf, ct*16 + m).
  // Row L2-norm: reduce squares across 16-lane half (covers all 64 columns).
  #pragma unroll
  for (int v = 0; v < 8; ++v) {
    float o0 = acc[0][v] + bias[0];
    float o1 = acc[1][v] + bias[1];
    float o2 = acc[2][v] + bias[2];
    float o3 = acc[3][v] + bias[3];
    float s = o0 * o0 + o1 * o1 + o2 * o2 + o3 * o3;
    s += __shfl_xor(s, 1);
    s += __shfl_xor(s, 2);
    s += __shfl_xor(s, 4);
    s += __shfl_xor(s, 8);
    const float scale = 1.0f / fmaxf(sqrtf(s), 1e-12f);
    const int r = row0 + v + 8 * khalf;
    if (r < n) {
      float* hr = h + (long long)r * TYPES;
      hr[ 0 + m] = 1.0f / (1.0f + __expf(-o0 * scale));
      hr[16 + m] = 1.0f / (1.0f + __expf(-o1 * scale));
      hr[32 + m] = 1.0f / (1.0f + __expf(-o2 * scale));
      hr[48 + m] = 1.0f / (1.0f + __expf(-o3 * scale));
    }
  }
}

// ------------------------------------ phase 3: label propagation (L2-resident)
__global__ void k_acc_init(const float* __restrict__ label, float* __restrict__ acc,
                           long long n64) {
  long long i = blockIdx.x * (long long)blockDim.x + threadIdx.x;
  if (i < n64) acc[i] = (1.0f - ALPHA) * label[i];
}

// one thread per (nonzero, feature): 64-thread groups move one 256B row coalesced
__global__ void k_prop_scatter(const int* __restrict__ prow, const int* __restrict__ pcol,
                               const float* __restrict__ pval, const float* __restrict__ h,
                               float* __restrict__ acc, long long pe) {
  long long gid = blockIdx.x * (long long)blockDim.x + threadIdx.x;
  if (gid >= pe * (long long)TYPES) return;
  int       f = (int)(gid & (TYPES - 1));
  long long i = gid >> 6;
  long long r = prow[i], c = pcol[i];
  float v = ALPHA * pval[i];
  atomicAdd(&acc[r * TYPES + f], v * h[c * TYPES + f]);
}

// ---------------------------------------------------------------- launcher
extern "C" void kernel_launch(void* const* d_in, const int* in_sizes, int n_in,
                              void* d_out, int out_size, void* d_ws, size_t ws_size,
                              hipStream_t stream) {
  (void)n_in; (void)ws_size;
  const float* x     = (const float*)d_in[0];
  const int*   edge  = (const int*)d_in[1];
  const int*   prow  = (const int*)d_in[2];
  const int*   pcol  = (const int*)d_in[3];
  const float* pval  = (const float*)d_in[4];
  const float* label = (const float*)d_in[5];
  const float* Wl    = (const float*)d_in[6];
  const float* bl    = (const float*)d_in[7];
  const float* Wr    = (const float*)d_in[8];

  const long long E      = in_sizes[1] / 2;
  const long long PE     = in_sizes[2];
  const int       N      = in_sizes[0] / F_IN;
  const long long TRUE_N = out_size / TYPES;

  const int* src = edge;          // edge_index[0]
  const int* dst = edge + E;      // edge_index[1]

  // workspace partition (floats), 256B-aligned segments
  float* ws  = (float*)d_ws;
  float* msg = ws;                                          // N*128
  float* deg = msg + (long long)N * F_IN;                   // N
  float* hA  = deg + (((long long)N + 63) & ~63LL);         // N*64
  float* hB  = hA + (long long)N * TYPES;                   // N*64

  const int B = 256;
  auto nb = [](long long n, int b) { return (unsigned)((n + b - 1) / b); };

  // phase 1: aggregation
  k_zero<<<nb((long long)N * F_IN, B), B, 0, stream>>>(msg, (long long)N * F_IN);
  k_zero<<<nb(N, B), B, 0, stream>>>(deg, N);
  k_deg<<<nb(E, B), B, 0, stream>>>(dst, deg, E);
  k_msg_scatter<<<nb(E * F_IN, B), B, 0, stream>>>(src, dst, x, msg, E);

  // phase 2: dual GEMM + normalize + sigmoid via WMMA (one wave per 16 rows)
  const long long waves = (N + 15) / 16;
  k_sage_wmma<<<nb(waves * 32, B), B, 0, stream>>>(msg, deg, x, Wl, bl, Wr, hA, N);

  // phase 3: 50 label-propagation rounds, double-buffered (stays in L2)
  float* cur = hA;
  float* nxt = hB;
  for (int it = 0; it < 50; ++it) {
    k_acc_init<<<nb((long long)N * TYPES, B), B, 0, stream>>>(label, nxt, (long long)N * TYPES);
    k_prop_scatter<<<nb(PE * TYPES, B), B, 0, stream>>>(prow, pcol, pval, cur, nxt, PE);
    float* t = cur; cur = nxt; nxt = t;
  }

  k_copy<<<nb(TRUE_N * TYPES, B), B, 0, stream>>>(cur, (float*)d_out, TRUE_N * TYPES);
}
